// SSD_61821759259084
// MI455X (gfx1250) — matrix operand order repaired and checked by reading, not compiled, not run on hardware
//
#include <hip/hip_runtime.h>
#include <math.h>

#define ATOTAL 8732
#define NIMG   32
#define MCAND  1200
#define TOPKN  400
#define NDETS  200
#define SORTN  16384
#define BBOX_CLIP 4.135166556742356f
#define IOU_THR 0.45f
#define CONF_THR 0.01f

typedef __attribute__((ext_vector_type(16))) _Float16 v16h;
typedef __attribute__((ext_vector_type(8)))  _Float16 v8h;
typedef __attribute__((ext_vector_type(8)))  float    v8f;

// ---------------------------------------------------------------------------
// Weight pre-pack: (OC, C, 3, 3) fp32 -> fp16 [tap][oc(pad 32)][C], zero pad.
// B fragment for lane (oc) is then two 8-half runs in K: 8h..8h+7, 16+8h..16+8h+7.
// ---------------------------------------------------------------------------
__global__ __launch_bounds__(256) void wt_pack_kernel(
    const float* __restrict__ wgt, _Float16* __restrict__ wT, int C, int OC)
{
    int i = blockIdx.x * 256 + threadIdx.x;          // over 9*32*C
    if (i >= 9 * 32 * C) return;
    int c   = i % C;
    int t2  = i / C;                                 // tap*32 + oc
    int oc  = t2 & 31;
    int tap = t2 >> 5;
    float v = 0.f;
    if (oc < OC) v = wgt[(oc * C + c) * 9 + tap];
    wT[i] = (_Float16)v;
}

// ---------------------------------------------------------------------------
// Activation pre-pack: (N, C, H, W) fp32 -> fp16 [n][H+2][W+2][C] with a
// zero halo, so every 3x3 tap read is in-bounds and contiguous in C.
// ---------------------------------------------------------------------------
__global__ __launch_bounds__(256) void xpack_kernel(
    const float* __restrict__ xin, _Float16* __restrict__ xp,
    int C, int H, int W)
{
    const int PH = H + 2, PW = W + 2;
    int i = blockIdx.x * 256 + threadIdx.x;          // over NIMG*PH*PW*C
    if (i >= NIMG * PH * PW * C) return;
    int c    = i % C;
    int rest = i / C;
    int px   = rest % PW; rest /= PW;
    int py   = rest % PH;
    int n    = rest / PH;
    int ix = px - 1, iy = py - 1;
    float v = 0.f;
    if (ix >= 0 && ix < W && iy >= 0 && iy < H)
        v = xin[((n * C + c) * H + iy) * W + ix];
    xp[i] = (_Float16)v;
}

// ---------------------------------------------------------------------------
// Implicit-GEMM 3x3 conv head, WMMA f32<=f16 16x16x32, K = tap*C + c.
// LDS-free: fragments loaded straight from pre-packed global buffers with
// b128 vector loads. Each wave: two 16-row M tiles x one 16-col OC tile.
// ---------------------------------------------------------------------------
__global__ __launch_bounds__(128) void conv_head_wmma(
    const _Float16* __restrict__ xp,    // [n][PH][PW][C]
    const _Float16* __restrict__ wT,    // [tap][oc32][C]
    const float* __restrict__ bias,     // (OC)
    float* __restrict__ outp,           // (N, ATOTAL, 4) base
    int C, int H, int W, int NAl, int levelOff, int Atotal)
{
    const int OC = NAl * 4;
    const int M  = H * W;
    const int PW = W + 2, PH = H + 2;
    const int wave = threadIdx.x >> 5;
    const int lane = threadIdx.x & 31;
    const int ln   = lane & 15;
    const int hf   = lane >> 4;
    const int mbase = (blockIdx.x * 4 + wave) * 32;  // 2 tiles of 16 rows
    const int oc0   = blockIdx.y * 16;
    const int n     = blockIdx.z;

    // staging rows for this lane (clamped; out-of-range rows masked at store)
    int m0 = mbase + ln;       if (m0 >= M) m0 = M - 1;
    int m1 = mbase + 16 + ln;  if (m1 >= M) m1 = M - 1;
    const int oy0 = m0 / W, ox0 = m0 - oy0 * W;
    const int oy1 = m1 / W, ox1 = m1 - oy1 * W;

    const _Float16* xb = xp + (size_t)n * PH * PW * C;
    const _Float16* wb = wT + (size_t)(oc0 + ln) * C + hf * 8;

    v8f acc0 = {0.f, 0.f, 0.f, 0.f, 0.f, 0.f, 0.f, 0.f};
    v8f acc1 = {0.f, 0.f, 0.f, 0.f, 0.f, 0.f, 0.f, 0.f};

    for (int tap = 0; tap < 9; ++tap) {
        const int fy = tap / 3, fx = tap - fy * 3;           // loop-scalar
        const _Float16* a0p = xb + (size_t)((oy0 + fy) * PW + (ox0 + fx)) * C + hf * 8;
        const _Float16* a1p = xb + (size_t)((oy1 + fy) * PW + (ox1 + fx)) * C + hf * 8;
        const _Float16* bp  = wb + (size_t)tap * 32 * C;

        #pragma unroll 2
        for (int c0 = 0; c0 < C; c0 += 32) {
            v8h aA0 = *(const v8h*)(a0p + c0);
            v8h aA1 = *(const v8h*)(a0p + c0 + 16);
            v8h aB0 = *(const v8h*)(a1p + c0);
            v8h aB1 = *(const v8h*)(a1p + c0 + 16);
            v8h b0  = *(const v8h*)(bp + c0);
            v8h b1  = *(const v8h*)(bp + c0 + 16);
            v16h afA = __builtin_shufflevector(aA0, aA1,
                0,1,2,3,4,5,6,7,8,9,10,11,12,13,14,15);
            v16h afB = __builtin_shufflevector(aB0, aB1,
                0,1,2,3,4,5,6,7,8,9,10,11,12,13,14,15);
            v16h bfr = __builtin_shufflevector(b0, b1,
                0,1,2,3,4,5,6,7,8,9,10,11,12,13,14,15);
            acc0 = __builtin_amdgcn_wmma_f32_16x16x32_f16(
                false, afA, false, bfr, (short)0, acc0, false, false);
            acc1 = __builtin_amdgcn_wmma_f32_16x16x32_f16(
                false, afB, false, bfr, (short)0, acc1, false, false);
        }
    }

    // epilogue: bias + scatter to (n, anchor, k) layout (true-m guards)
    const int oc = oc0 + ln;
    if (oc < OC) {
        const float bv = bias[oc];
        const int a = oc >> 2;
        const int k = oc & 3;
        #pragma unroll
        for (int r = 0; r < 8; ++r) {
            int m = mbase + r + hf * 8;
            if (m < M)
                outp[((n * Atotal + levelOff + m * NAl + a) << 2) + k] = acc0[r] + bv;
            m += 16;
            if (m < M)
                outp[((n * Atotal + levelOff + m * NAl + a) << 2) + k] = acc1[r] + bv;
        }
    }
}

// ---------------------------------------------------------------------------
// Decode boxes + softmax + confidence floor.
// ---------------------------------------------------------------------------
__global__ __launch_bounds__(256) void decode_kernel(
    const float* __restrict__ breg, const float* __restrict__ clog,
    const float* __restrict__ anchors,
    float* __restrict__ boxes, float* __restrict__ scores)
{
    int i = blockIdx.x * blockDim.x + threadIdx.x;
    if (i >= NIMG * ATOTAL) return;
    int img = i / ATOTAL;
    int a   = i - img * ATOTAL;

    float ax0 = anchors[a * 4 + 0], ay0 = anchors[a * 4 + 1];
    float ax1 = anchors[a * 4 + 2], ay1 = anchors[a * 4 + 3];
    float wdt = ax1 - ax0, hgt = ay1 - ay0;
    float cx = ax0 + 0.5f * wdt, cy = ay0 + 0.5f * hgt;

    const float* r = breg + (size_t)i * 4;
    float dx = r[0] * 0.1f, dy = r[1] * 0.1f;
    float dw = fminf(r[2] * 0.2f, BBOX_CLIP);
    float dh = fminf(r[3] * 0.2f, BBOX_CLIP);
    float pcx = dx * wdt + cx, pcy = dy * hgt + cy;
    float pw = expf(dw) * wdt, ph = expf(dh) * hgt;

    boxes[(size_t)i * 4 + 0] = fminf(fmaxf(pcx - 0.5f * pw, 0.f), 300.f);
    boxes[(size_t)i * 4 + 1] = fminf(fmaxf(pcy - 0.5f * ph, 0.f), 300.f);
    boxes[(size_t)i * 4 + 2] = fminf(fmaxf(pcx + 0.5f * pw, 0.f), 300.f);
    boxes[(size_t)i * 4 + 3] = fminf(fmaxf(pcy + 0.5f * ph, 0.f), 300.f);

    const float* lg = clog + (size_t)i * 4;
    float l0 = lg[0], l1 = lg[1], l2 = lg[2], l3 = lg[3];
    float mx = fmaxf(fmaxf(l0, l1), fmaxf(l2, l3));
    float e0 = expf(l0 - mx), e1 = expf(l1 - mx);
    float e2 = expf(l2 - mx), e3 = expf(l3 - mx);
    float inv = 1.f / (e0 + e1 + e2 + e3);
    float s1 = e1 * inv, s2 = e2 * inv, s3 = e3 * inv;
    scores[((size_t)img * 3 + 0) * ATOTAL + a] = (s1 > CONF_THR) ? s1 : 0.f;
    scores[((size_t)img * 3 + 1) * ATOTAL + a] = (s2 > CONF_THR) ? s2 : 0.f;
    scores[((size_t)img * 3 + 2) * ATOTAL + a] = (s3 > CONF_THR) ? s3 : 0.f;
}

// ---------------------------------------------------------------------------
// Per-(image,class) descending bitonic sort of packed keys in global scratch.
// key = (score_bits << 32) | (0xFFFFFFFF - anchor): lax.top_k tie rule.
// ---------------------------------------------------------------------------
__global__ __launch_bounds__(256) void topk_sort_kernel(
    const float* __restrict__ scores, unsigned long long* __restrict__ keys)
{
    int inst = blockIdx.x;                         // img*3 + class
    unsigned long long* kb = keys + (size_t)inst * SORTN;
    const float* sc = scores + (size_t)inst * ATOTAL;

    for (int j = threadIdx.x; j < SORTN; j += 256) {
        unsigned long long key = 0ull;
        if (j < ATOTAL) {
            unsigned int sb = __float_as_uint(sc[j]);   // scores >= 0
            key = ((unsigned long long)sb << 32) |
                  (unsigned long long)(0xFFFFFFFFu - (unsigned)j);
        }
        kb[j] = key;
    }
    __syncthreads();

    for (unsigned int k = 2; k <= SORTN; k <<= 1) {
        for (unsigned int j = k >> 1; j > 0; j >>= 1) {
            for (unsigned int i = threadIdx.x; i < SORTN; i += 256) {
                unsigned int l = i ^ j;
                if (l > i) {
                    unsigned long long a = kb[i], c = kb[l];
                    bool descSeg = ((i & k) == 0);
                    if (descSeg ? (a < c) : (a > c)) { kb[i] = c; kb[l] = a; }
                }
            }
            __syncthreads();
        }
    }
}

// ---------------------------------------------------------------------------
// Per-image: gather 3x400 candidates, LDS bitonic re-sort, sequential NMS
// on class-offset boxes, two-pass emit of 200 detections.
// ---------------------------------------------------------------------------
__global__ __launch_bounds__(256) void nms_kernel(
    const unsigned long long* __restrict__ keys,
    const float* __restrict__ boxes, float* __restrict__ out)
{
    __shared__ float cb[MCAND][4];               // raw candidate boxes
    __shared__ int   clbl[MCAND];                // label 1..3
    __shared__ unsigned long long skey[2048];    // (score_bits<<32)|(~pos)
    __shared__ int   keep[MCAND];

    const int img = blockIdx.x;

    for (int idx = threadIdx.x; idx < 2048; idx += 256) {
        if (idx < MCAND) {
            int c = idx / TOPKN;
            int r = idx - c * TOPKN;
            unsigned long long key = keys[((size_t)img * 3 + c) * SORTN + r];
            unsigned int anchor = 0xFFFFFFFFu - (unsigned int)(key & 0xFFFFFFFFull);
            const float* bp = boxes + ((size_t)img * ATOTAL + anchor) * 4;
            cb[idx][0] = bp[0]; cb[idx][1] = bp[1];
            cb[idx][2] = bp[2]; cb[idx][3] = bp[3];
            clbl[idx] = c + 1;
            keep[idx] = 1;
            skey[idx] = (key & 0xFFFFFFFF00000000ull) |
                        (unsigned long long)(0xFFFFFFFFu - (unsigned)idx);
        } else {
            skey[idx] = 0ull;
        }
    }
    __syncthreads();

    for (unsigned int k = 2; k <= 2048; k <<= 1) {
        for (unsigned int j = k >> 1; j > 0; j >>= 1) {
            for (unsigned int i = threadIdx.x; i < 2048; i += 256) {
                unsigned int l = i ^ j;
                if (l > i) {
                    unsigned long long a = skey[i], c = skey[l];
                    bool descSeg = ((i & k) == 0);
                    if (descSeg ? (a < c) : (a > c)) { skey[i] = c; skey[l] = a; }
                }
            }
            __syncthreads();
        }
    }

    for (int i = 0; i < MCAND; ++i) {
        __syncthreads();
        if (!keep[i]) continue;                       // uniform across block
        int pi = (int)(0xFFFFFFFFu - (unsigned int)(skey[i] & 0xFFFFFFFFull));
        float offi = 301.f * (float)clbl[pi];
        float ax0 = cb[pi][0] + offi, ay0 = cb[pi][1] + offi;
        float ax1 = cb[pi][2] + offi, ay1 = cb[pi][3] + offi;
        float areaA = (ax1 - ax0) * (ay1 - ay0);
        for (int j = i + 1 + (int)threadIdx.x; j < MCAND; j += 256) {
            if (!keep[j]) continue;
            int pj = (int)(0xFFFFFFFFu - (unsigned int)(skey[j] & 0xFFFFFFFFull));
            float offj = 301.f * (float)clbl[pj];
            float bx0 = cb[pj][0] + offj, by0 = cb[pj][1] + offj;
            float bx1 = cb[pj][2] + offj, by1 = cb[pj][3] + offj;
            float areaB = (bx1 - bx0) * (by1 - by0);
            float lx = fmaxf(ax0, bx0), ly = fmaxf(ay0, by0);
            float rx = fminf(ax1, bx1), ry = fminf(ay1, by1);
            float iw = fmaxf(rx - lx, 0.f), ih = fmaxf(ry - ly, 0.f);
            float inter = iw * ih;
            float iou = inter / (areaA + areaB - inter + 1e-9f);
            if (iou > IOU_THR) keep[j] = 0;
        }
    }
    __syncthreads();

    if (threadIdx.x == 0) {
        float* o = out + (size_t)img * NDETS * 6;
        int slot = 0;
        for (int pass = 0; pass < 2 && slot < NDETS; ++pass) {
            for (int m = 0; m < MCAND && slot < NDETS; ++m) {
                unsigned long long kk = skey[m];
                float s = __uint_as_float((unsigned int)(kk >> 32));
                int pm = (int)(0xFFFFFFFFu - (unsigned int)(kk & 0xFFFFFFFFull));
                bool valid = (keep[m] != 0) && (s > 0.f);
                if ((pass == 0) != valid) continue;
                o[slot * 6 + 0] = cb[pm][0];
                o[slot * 6 + 1] = cb[pm][1];
                o[slot * 6 + 2] = cb[pm][2];
                o[slot * 6 + 3] = cb[pm][3];
                o[slot * 6 + 4] = valid ? s : 0.f;
                o[slot * 6 + 5] = (float)clbl[pm];
                ++slot;
            }
        }
    }
}

// ---------------------------------------------------------------------------
extern "C" void kernel_launch(void* const* d_in, const int* in_sizes, int n_in,
                              void* d_out, int out_size, void* d_ws, size_t ws_size,
                              hipStream_t stream) {
    (void)in_sizes; (void)n_in; (void)out_size; (void)ws_size;

    static const int CH[6]  = {512, 1024, 512, 256, 256, 256};
    static const int SZ[6]  = {38, 19, 10, 5, 3, 1};
    static const int NAL[6] = {4, 6, 6, 6, 4, 4};
    int off[6]; int acc = 0;
    for (int l = 0; l < 6; ++l) { off[l] = acc; acc += NAL[l] * SZ[l] * SZ[l]; }

    const float* feats[6]; const float* regw[6]; const float* regb[6];
    const float* clsw[6];  const float* clsb[6];
    for (int l = 0; l < 6; ++l) {
        feats[l] = (const float*)d_in[l];
        regw[l]  = (const float*)d_in[6 + l];
        regb[l]  = (const float*)d_in[12 + l];
        clsw[l]  = (const float*)d_in[18 + l];
        clsb[l]  = (const float*)d_in[24 + l];
    }
    const float* anchors = (const float*)d_in[30];

    char* wsp = (char*)d_ws;
    auto carve = [&](size_t bytes) -> char* {
        char* p = wsp; wsp += (bytes + 255) & ~(size_t)255; return p;
    };
    float* breg   = (float*)carve((size_t)NIMG * ATOTAL * 4 * sizeof(float));
    float* clog   = (float*)carve((size_t)NIMG * ATOTAL * 4 * sizeof(float));
    float* boxes  = (float*)carve((size_t)NIMG * ATOTAL * 4 * sizeof(float));
    float* scores = (float*)carve((size_t)NIMG * 3 * ATOTAL * sizeof(float));
    unsigned long long* keys =
        (unsigned long long*)carve((size_t)NIMG * 3 * SORTN * sizeof(unsigned long long));
    _Float16* wTreg[6]; _Float16* wTcls[6];
    for (int l = 0; l < 6; ++l) {
        wTreg[l] = (_Float16*)carve((size_t)9 * 32 * CH[l] * sizeof(_Float16));
        wTcls[l] = (_Float16*)carve((size_t)9 * 32 * CH[l] * sizeof(_Float16));
    }
    // single padded-activation buffer, reused level by level (stream-ordered);
    // sized for level 0: 32 * 40 * 40 * 512 halfs
    _Float16* xp = (_Float16*)carve((size_t)NIMG * 40 * 40 * 512 * sizeof(_Float16));

    // weight pre-pack (fp32 OIHW -> fp16 [tap][oc32][C])
    for (int l = 0; l < 6; ++l) {
        int tot = 9 * 32 * CH[l];
        int OC = NAL[l] * 4;
        wt_pack_kernel<<<(tot + 255) / 256, 256, 0, stream>>>(regw[l], wTreg[l], CH[l], OC);
        wt_pack_kernel<<<(tot + 255) / 256, 256, 0, stream>>>(clsw[l], wTcls[l], CH[l], OC);
    }

    // per level: pack activations (halo + f16 + channel-last), then both heads
    for (int l = 0; l < 6; ++l) {
        int H = SZ[l], W = SZ[l], C = CH[l];
        int M  = H * W;
        int OC = NAL[l] * 4;
        int xtot = NIMG * (H + 2) * (W + 2) * C;
        xpack_kernel<<<(xtot + 255) / 256, 256, 0, stream>>>(feats[l], xp, C, H, W);
        dim3 grid((M + 127) / 128, (OC + 15) / 16, NIMG);
        conv_head_wmma<<<grid, 128, 0, stream>>>(
            xp, wTreg[l], regb[l], breg, C, H, W, NAL[l], off[l], ATOTAL);
        conv_head_wmma<<<grid, 128, 0, stream>>>(
            xp, wTcls[l], clsb[l], clog, C, H, W, NAL[l], off[l], ATOTAL);
    }

    int tot = NIMG * ATOTAL;
    decode_kernel<<<(tot + 255) / 256, 256, 0, stream>>>(breg, clog, anchors, boxes, scores);
    topk_sort_kernel<<<NIMG * 3, 256, 0, stream>>>(scores, keys);
    nms_kernel<<<NIMG, 256, 0, stream>>>(keys, boxes, (float*)d_out);
}
